// SelfAttention_37134287241830
// MI455X (gfx1250) — matrix operand order repaired and checked by reading, not compiled, not run on hardware
//
#include <hip/hip_runtime.h>
#include <hip/hip_bf16.h>

// ---------------------------------------------------------------------------
// Self-attention layer (B=2, T=2048, D=2048, NQ=16, NKV=4, H=128, RoPE, causal
// GQA) for MI455X / gfx1250.  All GEMM-shaped math on v_wmma_f32_16x16x32_bf16
// (wave32 WMMA, f32 accumulate).  K/V tiles staged via double-buffered
// global_load_async_to_lds_b128 DMA.
// ---------------------------------------------------------------------------

#define Bx 2
#define Tx 2048
#define Dx 2048
#define NQx 16
#define NKVx 4
#define Hx 128
#define Mx (Bx * Tx)            // 4096 global rows (b,t)

using u16 = unsigned short;
typedef __bf16 bf16_t;
typedef bf16_t v8bf  __attribute__((ext_vector_type(8)));   // 16 bytes
typedef bf16_t v16bf __attribute__((ext_vector_type(16)));  // 32 bytes (A/B frag)
typedef float  v8f   __attribute__((ext_vector_type(8)));   // C/D frag

// ---- workspace layout (bytes) ---------------------------------------------
#define OFF_XB   ((size_t)0)                       // X bf16      (4096,2048)
#define OFF_WQT  ((size_t)16777216)                // Wq^T bf16   (2048,2048)
#define OFF_WKT  ((size_t)25165824)                // Wk^T bf16   ( 512,2048)
#define OFF_WVT  ((size_t)27262976)                // Wv^T bf16   ( 512,2048)
#define OFF_WOT  ((size_t)29360128)                // Wo^T bf16   (2048,2048)
#define OFF_Q    ((size_t)37748736)                // q rope bf16 (B,NQ,T,H)
#define OFF_K    ((size_t)54525952)                // k rope bf16 (B,NKV,T,H)
#define OFF_VT   ((size_t)58720256)                // v^T   bf16 (B,NKV,H,T)
#define OFF_O    ((size_t)62914560)                // attn out bf16 (B,T,NQ,H)
// total ~79.7 MB

// ---- helpers ---------------------------------------------------------------
static __device__ inline u16 f2bf(float f) {
  unsigned u = __float_as_uint(f);
  unsigned r = (u + 0x7FFFu + ((u >> 16) & 1u)) >> 16;   // round-nearest-even
  return (u16)r;
}
static __device__ inline v16bf cat16(v8bf lo, v8bf hi) {
  return __builtin_shufflevector(lo, hi, 0,1,2,3,4,5,6,7,8,9,10,11,12,13,14,15);
}
static __device__ inline v8bf ld8(const u16* p) {
  return *reinterpret_cast<const v8bf*>(p);
}
static __device__ inline v8f wmma_bf16(v16bf a, v16bf b, v8f c) {
  return __builtin_amdgcn_wmma_f32_16x16x32_bf16(false, a, false, b, (short)0, c,
                                                 false, false);
}
static __device__ inline float rmax16(float v) {
  v = fmaxf(v, __shfl_xor(v, 1, 16));
  v = fmaxf(v, __shfl_xor(v, 2, 16));
  v = fmaxf(v, __shfl_xor(v, 4, 16));
  v = fmaxf(v, __shfl_xor(v, 8, 16));
  return v;
}
static __device__ inline float rsum16(float v) {
  v += __shfl_xor(v, 1, 16);
  v += __shfl_xor(v, 2, 16);
  v += __shfl_xor(v, 4, 16);
  v += __shfl_xor(v, 8, 16);
  return v;
}
// LDS byte-offset of a __shared__ object (addrspace(3) ptrtoint)
static __device__ inline unsigned lds_off(const void* p) {
  return (unsigned)(unsigned long long)(__attribute__((address_space(3))) const char*)p;
}
// one 16-byte async DMA: LDS[lds] = MEM[g]   (ASYNCcnt-tracked)
static __device__ inline void async_ld16(unsigned lds, const void* g) {
  asm volatile("global_load_async_to_lds_b128 %0, %1, off"
               :: "v"(lds), "v"((unsigned long long)g) : "memory");
}

// ---- prep: fp32 -> bf16 ----------------------------------------------------
__global__ __launch_bounds__(256) void cvt_bf16_kernel(const float* __restrict__ in,
                                                       u16* __restrict__ out, int n) {
  int i = blockIdx.x * 256 + threadIdx.x;
  if (i < n) out[i] = f2bf(in[i]);
}
// out[c*R + r] = bf16(in[r*C + c])
__global__ __launch_bounds__(256) void transpose_bf16_kernel(const float* __restrict__ in,
                                                             u16* __restrict__ out,
                                                             int R, int C) {
  int i = blockIdx.x * 256 + threadIdx.x;
  if (i >= R * C) return;
  int r = i / C, c = i - r * C;
  out[(size_t)c * R + r] = f2bf(in[(size_t)r * C + c]);
}

// ---------------------------------------------------------------------------
// Stage 1: QKV projection + RoPE.  One wave: 32 rows x 128 cols (full head) --
// two A fragments share every B fragment (16 WMMA per 20 b128 loads).
// B fragments pipelined 2 ahead, A fragments one k-step ahead.
// ---------------------------------------------------------------------------
__global__ __launch_bounds__(256) void qkv_rope_kernel(
    const u16* __restrict__ Xb, const u16* __restrict__ WqT,
    const u16* __restrict__ WkT, const u16* __restrict__ WvT,
    const int* __restrict__ qpos, u16* __restrict__ Qr, u16* __restrict__ Kr,
    u16* __restrict__ VT) {
  const int wave = threadIdx.x >> 5;
  const int lane = threadIdx.x & 31;
  const int n  = lane & 15;
  const int h8 = lane >> 4;
  const int task = blockIdx.x * 8 + wave;   // 128 row-tiles(32) * 24 heads
  const int rt = task / 24;
  const int hd = task - rt * 24;

  const u16* W;
  int nbase;
  if (hd < NQx)             { W = WqT; nbase = hd * Hx; }
  else if (hd < NQx + NKVx) { W = WkT; nbase = (hd - NQx) * Hx; }
  else                      { W = WvT; nbase = (hd - NQx - NKVx) * Hx; }

  const u16* arow0 = Xb + (size_t)(rt * 32 + n) * Dx;
  const u16* arow1 = arow0 + (size_t)16 * Dx;
  const u16* bbase = W + (size_t)(nbase + n) * Dx;

  v8f acc[2][8];
#pragma unroll
  for (int m = 0; m < 2; ++m)
#pragma unroll
    for (int j = 0; j < 8; ++j) acc[m][j] = (v8f)0.f;

  // pipeline prologue
  v16bf an0 = cat16(ld8(arow0 + h8 * 16), ld8(arow0 + h8 * 16 + 8));
  v16bf an1 = cat16(ld8(arow1 + h8 * 16), ld8(arow1 + h8 * 16 + 8));
  v16bf bpipe[2];
#pragma unroll
  for (int j = 0; j < 2; ++j) {
    const u16* bp = bbase + (size_t)j * 16 * Dx + h8 * 16;
    bpipe[j] = cat16(ld8(bp), ld8(bp + 8));
  }

#pragma unroll 1
  for (int k0 = 0; k0 < Dx; k0 += 32) {
    const int kn = (k0 + 32 < Dx) ? k0 + 32 : 0;   // next k (dummy on last iter)
    __builtin_prefetch(arow0 + k0 + 128, 0, 2);
    __builtin_prefetch(bbase + k0 + 128, 0, 2);
    v16bf a0 = an0, a1 = an1;
    an0 = cat16(ld8(arow0 + kn + h8 * 16), ld8(arow0 + kn + h8 * 16 + 8));
    an1 = cat16(ld8(arow1 + kn + h8 * 16), ld8(arow1 + kn + h8 * 16 + 8));
#pragma unroll
    for (int j = 0; j < 8; ++j) {
      v16bf b = bpipe[j & 1];
      const int jn = (j < 6) ? j + 2 : j - 6;       // fragment 2 ahead
      const int kj = (j < 6) ? k0 : kn;
      const u16* bp = bbase + (size_t)jn * 16 * Dx + kj + h8 * 16;
      bpipe[j & 1] = cat16(ld8(bp), ld8(bp + 8));
      acc[0][j] = wmma_bf16(a0, b, acc[0][j]);
      acc[1][j] = wmma_bf16(a1, b, acc[1][j]);
    }
  }

  // ---- epilogue: D-tile rows = m*16 + h8*8 + r, cols = j*16 + n ------------
  if (hd < NQx + NKVx) {
    float inv_ts[4];
#pragma unroll
    for (int j = 0; j < 4; ++j)
      inv_ts[j] = __expf(-(float)(j * 16 + n) * (9.210340371976184f / 64.f));
    const bool isq = hd < NQx;
    const int head = isq ? hd : hd - NQx;
    const int nheads = isq ? NQx : NKVx;
    u16* dst = isq ? Qr : Kr;
#pragma unroll
    for (int m = 0; m < 2; ++m)
#pragma unroll
      for (int r = 0; r < 8; ++r) {
        int g = rt * 32 + m * 16 + h8 * 8 + r;
        int bidx = g >> 11, t = g & (Tx - 1);
        float pos = (float)qpos[g];
        size_t base = (((size_t)bidx * nheads + head) * Tx + t) * Hx;
#pragma unroll
        for (int j = 0; j < 4; ++j) {
          float sn, cs;
          __sincosf(pos * inv_ts[j], &sn, &cs);
          float x1 = acc[m][j][r], x2 = acc[m][j + 4][r];
          dst[base + j * 16 + n]      = f2bf(x1 * cs - x2 * sn);
          dst[base + 64 + j * 16 + n] = f2bf(x2 * cs + x1 * sn);
        }
      }
  } else {
    const int kv = hd - NQx - NKVx;
#pragma unroll
    for (int m = 0; m < 2; ++m)
#pragma unroll
      for (int r = 0; r < 8; ++r) {
        int g = rt * 32 + m * 16 + h8 * 8 + r;
        int bidx = g >> 11, t = g & (Tx - 1);
        size_t hb = ((size_t)bidx * NKVx + kv) * Hx;
#pragma unroll
        for (int j = 0; j < 8; ++j)
          VT[(hb + j * 16 + n) * Tx + t] = f2bf(acc[m][j][r]);
      }
  }
}

// ---------------------------------------------------------------------------
// Stage 2: causal flash attention.  Block = 4 waves = 64 query rows of ONE
// head.  K/V 32-key chunks are staged into DOUBLE-BUFFERED LDS with
// global_load_async_to_lds_b128; chunk i+1's DMA overlaps chunk i's compute
// (async loads retire in order, so s_wait_asynccnt 0x8 releases chunk i while
// chunk i+1's 8 per-thread DMAs stay in flight).  Leading waves run extra
// fully-masked chunks (exact no-ops in online softmax) for uniform barriers.
// ---------------------------------------------------------------------------
__global__ __launch_bounds__(128) void flash_attn_kernel(
    const u16* __restrict__ Qr, const u16* __restrict__ Kr,
    const u16* __restrict__ VT, u16* __restrict__ Ob) {
  __shared__ u16 Kls[2][32 * Hx];   // [par][s_local][h]   2 x 8 KB
  __shared__ u16 Vls[2][Hx * 32];   // [par][h][s_local]   2 x 8 KB
  __shared__ u16 Pls[4][16 * 32];   // per-wave P re-stripe, 1 KB each

  const int tid  = threadIdx.x;
  const int wave = tid >> 5;
  const int lane = tid & 31;
  const int n  = lane & 15;
  const int h8 = lane >> 4;
  const int head = blockIdx.x >> 5;      // 0..31
  const int qtb  = blockIdx.x & 31;      // 64-row q block
  const int bidx = head >> 4, nq = head & 15, kv = nq >> 2;  // G = 4
  const int t0 = qtb * 64 + wave * 16;   // this wave's 16 q rows

  const u16* qrow = Qr + (((size_t)(bidx * NQx + nq) * Tx) + t0 + n) * Hx;
  const u16* Kb   = Kr + ((size_t)(bidx * NKVx + kv) * Tx) * Hx;
  const u16* VTb  = VT + ((size_t)(bidx * NKVx + kv) * Hx) * Tx;
  u16* myP = &Pls[wave][0];
  const unsigned klsb = lds_off(&Kls[0][0]);
  const unsigned vlsb = lds_off(&Vls[0][0]);

  // issue one 32-key chunk's DMA: 8 x 16B per thread (4 K + 4 V)
  auto issue = [&](int sb, int par) {
    const unsigned kb = klsb + par * (32 * Hx * 2);
    const unsigned vb = vlsb + par * (Hx * 32 * 2);
#pragma unroll
    for (int p = 0; p < 4; ++p) {        // K: 512 16B units  [s][h]
      int u = tid + p * 128;
      int row = u >> 4, c = u & 15;
      async_ld16(kb + u * 16,
                 (const char*)Kb + ((size_t)(sb + row) * Hx + c * 8) * 2);
    }
#pragma unroll
    for (int p = 0; p < 4; ++p) {        // V: 512 16B units  [h][s]
      int u = tid + p * 128;
      int h = u >> 2, c = u & 3;
      async_ld16(vb + u * 16,
                 (const char*)VTb + ((size_t)h * Tx + sb + c * 8) * 2);
    }
  };

  v16bf qf[4];
#pragma unroll
  for (int hk = 0; hk < 4; ++hk)
    qf[hk] = cat16(ld8(qrow + hk * 32 + h8 * 16), ld8(qrow + hk * 32 + h8 * 16 + 8));

  v8f o[8];
#pragma unroll
  for (int j = 0; j < 8; ++j) o[j] = (v8f)0.f;
  float m_st[8], l_st[8];
#pragma unroll
  for (int r = 0; r < 8; ++r) { m_st[r] = -1e30f; l_st[r] = 0.f; }

  const int nch = (qtb * 64 + 64) >> 5;  // uniform chunk count per block
  issue(0, 0);                           // prologue DMA
  for (int ch = 0; ch < nch; ++ch) {
    const int sb = ch * 32, par = ch & 1;
    if (ch + 1 < nch) {
      issue(sb + 32, par ^ 1);                           // overlap next DMA
      asm volatile("s_wait_asynccnt 0x8" ::: "memory");  // chunk ch resident
    } else {
      asm volatile("s_wait_asynccnt 0x0" ::: "memory");
    }
    __syncthreads();                     // chunk visible to all waves
    const u16* Kc = &Kls[par][0];
    const u16* Vc = &Vls[par][0];

    // ---- S = Q @ K^T (two 16x16 tiles from LDS) ---------------------------
    v8f s0 = (v8f)0.f, s1 = (v8f)0.f;
    const u16* k0p = Kc + (size_t)n * Hx;
    const u16* k1p = Kc + (size_t)(16 + n) * Hx;
#pragma unroll
    for (int hk = 0; hk < 4; ++hk) {
      v16bf kf0 = cat16(ld8(k0p + hk * 32 + h8 * 16), ld8(k0p + hk * 32 + h8 * 16 + 8));
      v16bf kf1 = cat16(ld8(k1p + hk * 32 + h8 * 16), ld8(k1p + hk * 32 + h8 * 16 + 8));
      s0 = wmma_bf16(qf[hk], kf0, s0);
      s1 = wmma_bf16(qf[hk], kf1, s1);
    }
    // ---- causal mask ------------------------------------------------------
    if (sb + 31 > t0) {
#pragma unroll
      for (int r = 0; r < 8; ++r) {
        int t = t0 + h8 * 8 + r;
        if (sb + n > t)      s0[r] = -1e30f;
        if (sb + 16 + n > t) s1[r] = -1e30f;
      }
    }
    // ---- online softmax (rows across 16 lanes of a half) ------------------
    float alpha[8];
#pragma unroll
    for (int r = 0; r < 8; ++r) {
      float cm = rmax16(fmaxf(s0[r], s1[r]));
      float mn = fmaxf(m_st[r], cm);
      float al = __expf(m_st[r] - mn);
      float p0 = __expf(s0[r] - mn);
      float p1 = __expf(s1[r] - mn);
      l_st[r] = l_st[r] * al + rsum16(p0 + p1);
      m_st[r] = mn;
      alpha[r] = al;
      int row = h8 * 8 + r;
      myP[row * 32 + n]      = f2bf(p0);
      myP[row * 32 + 16 + n] = f2bf(p1);
    }
#pragma unroll
    for (int j = 0; j < 8; ++j)
#pragma unroll
      for (int r = 0; r < 8; ++r) o[j][r] *= alpha[r];

    asm volatile("s_wait_dscnt 0" ::: "memory");   // P visible within wave
    const u16* prow = myP + n * 32;
    v16bf pf = cat16(ld8(prow + h8 * 16), ld8(prow + h8 * 16 + 8));

    // ---- O += P @ V (V fragments from LDS) --------------------------------
#pragma unroll
    for (int j = 0; j < 8; ++j) {
      const u16* vp = Vc + (size_t)(j * 16 + n) * 32 + h8 * 16;
      o[j] = wmma_bf16(pf, cat16(ld8(vp), ld8(vp + 8)), o[j]);
    }
    __syncthreads();                     // readers done before buffer reuse
  }

  // ---- normalize + store (b, t, nq, h) bf16 --------------------------------
  float invl[8];
#pragma unroll
  for (int r = 0; r < 8; ++r) invl[r] = 1.0f / l_st[r];
#pragma unroll
  for (int r = 0; r < 8; ++r) {
    int t = t0 + h8 * 8 + r;
    size_t base = (((size_t)bidx * Tx + t) * NQx + nq) * Hx;
#pragma unroll
    for (int j = 0; j < 8; ++j)
      Ob[base + j * 16 + n] = f2bf(o[j][r] * invl[r]);
  }
}

// ---------------------------------------------------------------------------
// Stage 3: output projection, 32x64 tile per wave, pipelined like stage 1.
// ---------------------------------------------------------------------------
__global__ __launch_bounds__(256) void oproj_kernel(const u16* __restrict__ Ob,
                                                    const u16* __restrict__ WoT,
                                                    float* __restrict__ out) {
  const int wave = threadIdx.x >> 5;
  const int lane = threadIdx.x & 31;
  const int n  = lane & 15;
  const int h8 = lane >> 4;
  const int task = blockIdx.x * 8 + wave;   // 128 row-tiles(32) * 32 col-groups
  const int rt = task >> 5;
  const int cg = task & 31;
  const int K = NQx * Hx;   // 2048

  const u16* arow0 = Ob + (size_t)(rt * 32 + n) * K;
  const u16* arow1 = arow0 + (size_t)16 * K;
  const u16* bbase = WoT + (size_t)(cg * 64 + n) * K;

  v8f acc[2][4];
#pragma unroll
  for (int m = 0; m < 2; ++m)
#pragma unroll
    for (int j = 0; j < 4; ++j) acc[m][j] = (v8f)0.f;

  v16bf an0 = cat16(ld8(arow0 + h8 * 16), ld8(arow0 + h8 * 16 + 8));
  v16bf an1 = cat16(ld8(arow1 + h8 * 16), ld8(arow1 + h8 * 16 + 8));
  v16bf bpipe[2];
#pragma unroll
  for (int j = 0; j < 2; ++j) {
    const u16* bp = bbase + (size_t)j * 16 * K + h8 * 16;
    bpipe[j] = cat16(ld8(bp), ld8(bp + 8));
  }

#pragma unroll 1
  for (int k0 = 0; k0 < K; k0 += 32) {
    const int kn = (k0 + 32 < K) ? k0 + 32 : 0;
    __builtin_prefetch(arow0 + k0 + 128, 0, 2);
    __builtin_prefetch(bbase + k0 + 128, 0, 2);
    v16bf a0 = an0, a1 = an1;
    an0 = cat16(ld8(arow0 + kn + h8 * 16), ld8(arow0 + kn + h8 * 16 + 8));
    an1 = cat16(ld8(arow1 + kn + h8 * 16), ld8(arow1 + kn + h8 * 16 + 8));
#pragma unroll
    for (int j = 0; j < 4; ++j) {
      v16bf b = bpipe[j & 1];
      const int jn = (j < 2) ? j + 2 : j - 2;
      const int kj = (j < 2) ? k0 : kn;
      const u16* bp = bbase + (size_t)jn * 16 * K + kj + h8 * 16;
      bpipe[j & 1] = cat16(ld8(bp), ld8(bp + 8));
      acc[0][j] = wmma_bf16(a0, b, acc[0][j]);
      acc[1][j] = wmma_bf16(a1, b, acc[1][j]);
    }
  }
#pragma unroll
  for (int m = 0; m < 2; ++m)
#pragma unroll
    for (int r = 0; r < 8; ++r) {
      size_t g = (size_t)(rt * 32 + m * 16 + h8 * 8 + r) * Dx + cg * 64;
#pragma unroll
      for (int j = 0; j < 4; ++j) out[g + j * 16 + n] = acc[m][j][r];
    }
}

// ---------------------------------------------------------------------------
extern "C" void kernel_launch(void* const* d_in, const int* in_sizes, int n_in,
                              void* d_out, int out_size, void* d_ws, size_t ws_size,
                              hipStream_t stream) {
  const float* X  = (const float*)d_in[0];
  const float* Wq = (const float*)d_in[1];
  const float* Wk = (const float*)d_in[2];
  const float* Wv = (const float*)d_in[3];
  const float* Wo = (const float*)d_in[4];
  const int* qpos = (const int*)d_in[5];
  float* out = (float*)d_out;

  char* ws = (char*)d_ws;
  u16* Xb  = (u16*)(ws + OFF_XB);
  u16* WqT = (u16*)(ws + OFF_WQT);
  u16* WkT = (u16*)(ws + OFF_WKT);
  u16* WvT = (u16*)(ws + OFF_WVT);
  u16* WoT = (u16*)(ws + OFF_WOT);
  u16* Qr  = (u16*)(ws + OFF_Q);
  u16* Kr  = (u16*)(ws + OFF_K);
  u16* VT  = (u16*)(ws + OFF_VT);
  u16* Ob  = (u16*)(ws + OFF_O);

  cvt_bf16_kernel<<<(Mx * Dx) / 256, 256, 0, stream>>>(X, Xb, Mx * Dx);
  transpose_bf16_kernel<<<(Dx * NQx * Hx) / 256, 256, 0, stream>>>(Wq, WqT, Dx, NQx * Hx);
  transpose_bf16_kernel<<<(Dx * NKVx * Hx) / 256, 256, 0, stream>>>(Wk, WkT, Dx, NKVx * Hx);
  transpose_bf16_kernel<<<(Dx * NKVx * Hx) / 256, 256, 0, stream>>>(Wv, WvT, Dx, NKVx * Hx);
  transpose_bf16_kernel<<<(NQx * Hx * Dx) / 256, 256, 0, stream>>>(Wo, WoT, NQx * Hx, Dx);

  // stage 1: QKV + RoPE   (128 row-tiles(32) * 24 heads) / 8 waves
  qkv_rope_kernel<<<384, 256, 0, stream>>>(Xb, WqT, WkT, WvT, qpos, Qr, Kr, VT);
  // stage 2: causal flash attention   (32 heads * 32 q-blocks(64))
  flash_attn_kernel<<<1024, 128, 0, stream>>>(Qr, Kr, VT, Ob);
  // stage 3: output projection   (128 row-tiles(32) * 32 col-groups(64)) / 8 waves
  oproj_kernel<<<512, 256, 0, stream>>>(Ob, WoT, out);
}
